// SymmetricContraction_63728724738605
// MI455X (gfx1250) — matrix operand order repaired and checked by reading, not compiled, not run on hardware
//
#include <hip/hip_runtime.h>
#include <utility>

typedef __attribute__((ext_vector_type(8)))  _Float16 v8h;
typedef __attribute__((ext_vector_type(16))) _Float16 v16h;
typedef __attribute__((ext_vector_type(8)))  float    v8f;

#define NN    32000
#define CC    64
#define SS    10
#define NPROJ 32
#define NMONO 219   // 9 linear + 45 quad + 165 cubic
#define KPAD  224   // 7 WMMA k-steps of 32
#define NCOL  40    // 4 u * 10 p
#define CPAD  48    // 3 WMMA col tiles of 16
#define AST   232   // A LDS row stride (halves), 16B-aligned rows
#define TST   49    // T LDS row stride (floats), odd -> conflict-free reads
#define NWAVES 4000 // 2000 blocks * 2 waves
#define NTILES 64000

// ---------------- compile-time monomial table ----------------
struct MonoTab { int a[KPAD]; int b[KPAD]; int c[KPAD]; int ar[KPAD]; };
constexpr MonoTab makeTab() {
  MonoTab t{};
  int m = 0;
  for (int i = 0; i < 9; i++) { t.a[m] = i; t.ar[m] = 1; m++; }
  for (int i = 0; i < 9; i++)
    for (int j = i; j < 9; j++) { t.a[m] = i; t.b[m] = j; t.ar[m] = 2; m++; }
  for (int i = 0; i < 9; i++)
    for (int j = i; j < 9; j++)
      for (int k = j; k < 9; k++) { t.a[m] = i; t.b[m] = j; t.c[m] = k; t.ar[m] = 3; m++; }
  return t; // entries 219..223 stay arity 0 -> zero pad
}
constexpr MonoTab MT = makeTab();

template <int M>
__device__ __forceinline__ _Float16 mono(const float* __restrict__ v) {
  if constexpr (MT.ar[M] == 1)      return (_Float16)v[MT.a[M]];
  else if constexpr (MT.ar[M] == 2) return (_Float16)(v[MT.a[M]] * v[MT.b[M]]);
  else if constexpr (MT.ar[M] == 3) return (_Float16)(v[MT.a[M]] * v[MT.b[M]] * v[MT.c[M]]);
  else                              return (_Float16)0.0f;
}

template <int T>
__device__ __forceinline__ void stageChunk(const float* __restrict__ v, _Float16* Arow) {
  v8h pk = { mono<T*8+0>(v), mono<T*8+1>(v), mono<T*8+2>(v), mono<T*8+3>(v),
             mono<T*8+4>(v), mono<T*8+5>(v), mono<T*8+6>(v), mono<T*8+7>(v) };
  *reinterpret_cast<v8h*>(Arow + T * 8) = pk;
}
template <int... Is>
__device__ __forceinline__ void stageAll(const float* __restrict__ v, _Float16* Arow,
                                         std::integer_sequence<int, Is...>) {
  (stageChunk<Is>(v, Arow), ...);
}

// ---------------- prep kernel 1: wpT[s][c][pg] = sum_a w[s,a,c]*proj[a,pg] ----------------
__global__ void sc_prep_w(const float* __restrict__ w, const float* __restrict__ proj,
                          float* __restrict__ wpT) {
  int t = blockIdx.x * blockDim.x + threadIdx.x; // t = ((s*CC)+c)*10 + pg
  if (t >= SS * CC * 10) return;
  int pg = t % 10;
  int c  = (t / 10) % CC;
  int s  = t / (10 * CC);
  float acc = 0.f;
  for (int a = 0; a < NPROJ; a++)
    acc += w[(s * NPROJ + a) * CC + c] * proj[a * 10 + pg];
  wpT[t] = acc;
}

// ---------------- prep kernel 2: symmetrized U, f16, col-major UtG[col][KPAD] ----------------
__global__ void sc_prep_u(const float* __restrict__ u1, const float* __restrict__ u2,
                          const float* __restrict__ u3, _Float16* __restrict__ UtG) {
  int t = blockIdx.x * blockDim.x + threadIdx.x; // t = col*KPAD + m
  if (t >= CPAD * KPAD) return;
  int m = t % KPAD;
  int col = t / KPAD;
  float val = 0.f;
  if (col < NCOL && m < NMONO) {
    int u = col / 10, pg = col % 10;
    if (m < 9) {                          // linear, pg in [0,2)
      if (pg < 2) val = u1[(u * 9 + m) * 2 + pg];
    } else if (m < 54) {                  // quadratic, pg in [2,6)
      if (pg >= 2 && pg < 6) {
        int mq = m - 9, i = 0, j = 0;
        for (i = 0; i < 9; i++) { int cnt = 9 - i; if (mq < cnt) { j = i + mq; break; } mq -= cnt; }
        int p = pg - 2;
        val = u2[((u * 9 + i) * 9 + j) * 4 + p];
        if (i != j) val += u2[((u * 9 + j) * 9 + i) * 4 + p];
      }
    } else {                              // cubic, pg in [6,10)
      if (pg >= 6) {
        int mc = m - 54, i = -1, j = -1, k = -1;
        for (int ii = 0; ii < 9 && i < 0; ii++)
          for (int jj = ii; jj < 9; jj++) {
            int cnt = 9 - jj;
            if (mc < cnt) { i = ii; j = jj; k = jj + mc; break; }
            mc -= cnt;
          }
        int p = pg - 6;
        auto U3 = [&](int A, int B, int C) {
          return u3[(((u * 9 + A) * 9 + B) * 9 + C) * 4 + p];
        };
        if (i == j && j == k)      val = U3(i, i, i);
        else if (i == j)           val = U3(i, i, k) + U3(i, k, i) + U3(k, i, i);
        else if (j == k)           val = U3(i, j, j) + U3(j, i, j) + U3(j, j, i);
        else                       val = U3(i, j, k) + U3(i, k, j) + U3(j, i, k)
                                       + U3(j, k, i) + U3(k, i, j) + U3(k, j, i);
      }
    }
  }
  UtG[t] = (_Float16)val;
}

// ---------------- main kernel ----------------
__global__ __launch_bounds__(64)
void sc_main(const float* __restrict__ x, const float* __restrict__ wpT,
             const _Float16* __restrict__ UtG, const int* __restrict__ cnts,
             float* __restrict__ out) {
  __shared__ __align__(16) _Float16 UtL[CPAD * KPAD];      // 21504 B
  __shared__ __align__(16) _Float16 Alds[2][32 * AST];     // 2 * 14848 B
  __shared__ int cum[SS];

  const int tid  = threadIdx.x;
  const int wid  = tid >> 5;
  const int lane = tid & 31;
  const int ln   = lane & 15;
  const int hf   = lane >> 4;

  // block-cooperative copy of U tile into LDS (1344 uint4)
  {
    const uint4* src = reinterpret_cast<const uint4*>(UtG);
    uint4*       dst = reinterpret_cast<uint4*>(UtL);
    for (int i = tid; i < (CPAD * KPAD) / 8; i += 64) dst[i] = src[i];
  }
  if (tid == 0) { int run = 0; for (int s = 0; s < SS; s++) { run += cnts[s]; cum[s] = run; } }
  __syncthreads();

  const int gwave = blockIdx.x * 2 + wid;

  for (int it = 0; it < NTILES / NWAVES; ++it) {
    const int tile = gwave + NWAVES * it;
    const int r    = tile * 32 + lane;          // global row = n*CC + c

    // load this row's 9 features
    float v[9];
    const float* xr = x + (size_t)r * 9;
    #pragma unroll
    for (int i = 0; i < 9; i++) v[i] = xr[i];

    // stage all 224 padded monomials for row==lane (compile-time indices, packed b128 stores)
    stageAll(v, &Alds[wid][lane * AST], std::make_integer_sequence<int, 28>{});
    asm volatile("" ::: "memory");

    v8f acc[2][3] = {};
    #pragma unroll
    for (int kk = 0; kk < 7; ++kk) {
      v16h a[2], b[3];
      #pragma unroll
      for (int rt = 0; rt < 2; rt++) {
        // ISA 16-bit A layout: half-wave 0 holds K{0..7,16..23}, half-wave 1 holds K{8..15,24..31}
        const _Float16* ap = &Alds[wid][(rt * 16 + ln) * AST + kk * 32 + hf * 8];
        v8h lo = *reinterpret_cast<const v8h*>(ap);
        v8h hi = *reinterpret_cast<const v8h*>(ap + 16);
        a[rt] = __builtin_shufflevector(lo, hi, 0,1,2,3,4,5,6,7,8,9,10,11,12,13,14,15);
      }
      #pragma unroll
      for (int ct = 0; ct < 3; ct++) {
        // B: lane n = col, half-wave 0 holds K{0..7,16..23}, half-wave 1 holds K{8..15,24..31}
        const _Float16* bp = &UtL[(ct * 16 + ln) * KPAD + kk * 32 + hf * 8];
        v8h lo = *reinterpret_cast<const v8h*>(bp);
        v8h hi = *reinterpret_cast<const v8h*>(bp + 16);
        b[ct] = __builtin_shufflevector(lo, hi, 0,1,2,3,4,5,6,7,8,9,10,11,12,13,14,15);
      }
      #pragma unroll
      for (int rt = 0; rt < 2; rt++)
        #pragma unroll
        for (int ct = 0; ct < 3; ct++)
          acc[rt][ct] = __builtin_amdgcn_wmma_f32_16x16x32_f16(
              false, a[rt], false, b[ct], (short)0, acc[rt][ct], false, false);
    }
    asm volatile("" ::: "memory");

    // park T = 32x48 f32 in LDS (overlay dead A-stage region)
    float* Tb = reinterpret_cast<float*>(&Alds[wid][0]);
    #pragma unroll
    for (int rt = 0; rt < 2; rt++)
      #pragma unroll
      for (int ct = 0; ct < 3; ct++)
        #pragma unroll
        for (int g = 0; g < 8; g++)
          Tb[(rt * 16 + hf * 8 + g) * TST + ct * 16 + ln] = acc[rt][ct][g];
    asm volatile("" ::: "memory");

    // epilogue: lane owns row==lane; contract 10 p-slots with per-(species,c) weights
    const int n = r >> 6, c = r & 63;
    int s = 0;
    #pragma unroll
    for (int q = 0; q < SS; q++) s += (n >= cum[q]) ? 1 : 0;
    const float* wv = wpT + (size_t)(s * CC + c) * 10;
    float wr[10];
    #pragma unroll
    for (int p = 0; p < 10; p++) wr[p] = wv[p];

    float o[4];
    #pragma unroll
    for (int u = 0; u < 4; u++) {
      float sum = 0.f;
      #pragma unroll
      for (int p = 0; p < 10; p++) sum += Tb[lane * TST + u * 10 + p] * wr[p];
      o[u] = sum;
    }
    float4 o4 = make_float4(o[0], o[1], o[2], o[3]);
    *reinterpret_cast<float4*>(out + (size_t)r * 4) = o4;
    asm volatile("" ::: "memory");
  }
}

extern "C" void kernel_launch(void* const* d_in, const int* in_sizes, int n_in,
                              void* d_out, int out_size, void* d_ws, size_t ws_size,
                              hipStream_t stream) {
  const float* x    = (const float*)d_in[0];
  const float* w    = (const float*)d_in[1];
  const float* proj = (const float*)d_in[2];
  const float* u1   = (const float*)d_in[3];
  const float* u2   = (const float*)d_in[4];
  const float* u3   = (const float*)d_in[5];
  const int*  cnts  = (const int*)d_in[6];
  float* out = (float*)d_out;

  float*    wpT = (float*)d_ws;                          // 6400 f32  = 25600 B
  _Float16* UtG = (_Float16*)((char*)d_ws + 25600);      // 10752 f16 = 21504 B (16B-aligned)

  sc_prep_w<<<25, 256, 0, stream>>>(w, proj, wpT);       // 6400 threads
  sc_prep_u<<<42, 256, 0, stream>>>(u1, u2, u3, UtG);    // 10752 threads
  sc_main  <<<2000, 64, 0, stream>>>(x, wpT, UtG, cnts, out);
}
